// MultiHeadSelfAttention_59150289600877
// MI455X (gfx1250) — matrix operand order repaired and checked by reading, not compiled, not run on hardware
//
#include <hip/hip_runtime.h>
#include <hip/hip_bf16.h>
#include <math.h>

// Problem constants (from reference): B=2, S=2048, D=1024, H=16, DK=64
#define B_  2
#define S_  2048
#define D_  1024
#define H_  16
#define DK_ 64

typedef __attribute__((ext_vector_type(16))) __bf16 v16bf;
typedef __attribute__((ext_vector_type(8)))  float  v8f;
typedef __attribute__((ext_vector_type(4)))  unsigned int v4u;
typedef __attribute__((ext_vector_type(8)))  int    v8i;
typedef __attribute__((ext_vector_type(4)))  int    v4i;

union Frag16 { v16bf v; uint4 q[2]; };

__device__ __forceinline__ v8f wmma_bf16(v16bf a, v16bf b, v8f c) {
  // D = A(16x32 bf16) * B(32x16 bf16) + C(16x16 f32)
  return __builtin_amdgcn_wmma_f32_16x16x32_bf16(false, a, false, b, (short)0, c,
                                                 false, false);
}

// A operand (16M x 32K) from row-major memory (each row: K contiguous).
// ISA layout: lane r (r<16) holds row r, K {kb..kb+7, kb+16..kb+23}, kb=0;
// lane r+16 holds row r with kb=8.
__device__ __forceinline__ v16bf load_a16x32(const __bf16* p0, int ld, int lane) {
  const int r  = lane & 15;
  const int kb = (lane >> 4) << 3;  // 0 or 8
  const __bf16* p = p0 + (size_t)r * ld + kb;
  Frag16 f;
  f.q[0] = *reinterpret_cast<const uint4*>(p);       // K kb..kb+7
  f.q[1] = *reinterpret_cast<const uint4*>(p + 16);  // K kb+16..kb+23
  return f.v;
}

// B operand (32K x 16N) where memory holds, per column n, K contiguous with
// stride ld between columns. ISA layout: lane n holds col n K {0..7, 16..23},
// lane n+16 holds col n K {8..15, 24..31}.
__device__ __forceinline__ v16bf load_b32x16(const __bf16* p0, int ld, int lane) {
  const int n  = lane & 15;
  const int kh = (lane >> 4) << 3;  // 0 or 8
  const __bf16* p = p0 + (size_t)n * ld + kh;
  Frag16 f;
  f.q[0] = *reinterpret_cast<const uint4*>(p);       // K kh..kh+7
  f.q[1] = *reinterpret_cast<const uint4*>(p + 16);  // K kh+16..kh+23
  return f.v;
}

__device__ __forceinline__ unsigned short f2bf(float f) {
  union { __bf16 h; unsigned short u; } c;
  c.h = (__bf16)f;
  return c.u;
}

// Cross-lane xor exchange within wave32 via LDS-permute hardware (no LDS mem).
#define SWZF(x, mask) \
  __int_as_float(__builtin_amdgcn_ds_swizzle(__float_as_int(x), ((mask) << 10) | 0x1f))

__device__ __forceinline__ float rowmax16(float x) {
  x = fmaxf(x, SWZF(x, 1));
  x = fmaxf(x, SWZF(x, 2));
  x = fmaxf(x, SWZF(x, 4));
  x = fmaxf(x, SWZF(x, 8));
  return x;
}
__device__ __forceinline__ float rowsum16(float x) {
  x += SWZF(x, 1);
  x += SWZF(x, 2);
  x += SWZF(x, 4);
  x += SWZF(x, 8);
  return x;
}

// ---------------------------------------------------------------------------
// Tensor Data Mover: async 2D tile load Global -> LDS.
// Builds D# group0/group1 per CDNA5 ISA:
//   g0: [1:0]=count=1, [63:32]=lds_addr, [120:64]=global_addr, [127:126]=2
//   g1: [17:16]=data_size=1 (2 bytes), tensor_dim0/1, tile_dim0/1,
//       tensor_dim0_stride; workgroup_mask=0 (not in a cluster)
// 2D tile => remaining descriptor groups zero-filled (unused dims).
// This toolchain exposes the 6-arg builtin:
//   (uint32x4 g0, int32x8 g1, int32x4, int32x4, int32x8, i32 cpol)
// Dimensions/strides are in elements (data_size units).
// ---------------------------------------------------------------------------
__device__ __forceinline__ void tdm_load_2d_bf16(const __bf16* gsrc,
                                                 unsigned lds_byte_off,
                                                 unsigned tensor_d0,
                                                 unsigned tensor_d1,
                                                 unsigned tile_d0,
                                                 unsigned tile_d1,
                                                 unsigned stride_d0) {
  const unsigned long long ga = (unsigned long long)(uintptr_t)gsrc;
  v4u g0;
  g0[0] = 1u;                                   // count=1 (valid), user bits 0
  g0[1] = lds_byte_off;                         // lds_addr (bytes)
  g0[2] = (unsigned)(ga & 0xffffffffu);         // global_addr[31:0]
  g0[3] = (unsigned)((ga >> 32) & 0x01ffffffu)  // global_addr[56:32]
          | (2u << 30);                         // type=2 ("image")
  v8i g1;
  g1[0] = (int)(1u << 16);                      // data_size=1 -> 2-byte elems
  g1[1] = (int)((tensor_d0 & 0xffffu) << 16);   // tensor_dim0[15:0]
  g1[2] = (int)(((tensor_d0 >> 16) & 0xffffu) | ((tensor_d1 & 0xffffu) << 16));
  g1[3] = (int)(((tensor_d1 >> 16) & 0xffffu) | ((tile_d0 & 0xffffu) << 16));
  g1[4] = (int)(tile_d1 & 0xffffu);             // tile_dim1; tile_dim2=0
  g1[5] = (int)stride_d0;                       // tensor_dim0_stride[31:0]
  g1[6] = 0;                                    // stride hi / dim1_stride lo
  g1[7] = 0;                                    // dim1_stride hi (unused, 2D)
  const v4i z4 = {0, 0, 0, 0};
  const v8i z8 = {0, 0, 0, 0, 0, 0, 0, 0};
  __builtin_amdgcn_tensor_load_to_lds(g0, g1, z4, z4, z8, 0);
}

// ---------------------------------------------------------------------------
// fp32 -> bf16 bulk conversion (vectorized: float4 in, 4x bf16 out)
// ---------------------------------------------------------------------------
__global__ __launch_bounds__(256) void cvt_f32_bf16(const float* __restrict__ src,
                                                    __bf16* __restrict__ dst,
                                                    int n4) {
  const int i = blockIdx.x * 256 + threadIdx.x;
  if (i >= n4) return;
  const float4 f = reinterpret_cast<const float4*>(src)[i];
  union { __bf16 h[4]; uint2 u; } c;
  c.h[0] = (__bf16)f.x; c.h[1] = (__bf16)f.y;
  c.h[2] = (__bf16)f.z; c.h[3] = (__bf16)f.w;
  reinterpret_cast<uint2*>(dst)[i] = c.u;
}

// ---------------------------------------------------------------------------
// QKV projection (C = X * W^T) with fused RoPE (Q,K) / transpose store (V).
// Block: 256 threads = 8 waves, tile 128(M) x 64(N); N-tile == one head.
// A (128x32) and B (64x32) K-step tiles are staged in LDS by the Tensor Data
// Mover (double buffered, wave 0 issues, s_wait_tensorcnt + barrier handoff).
// Wave: 32x32 output = 2x2 WMMA tiles, K loop step 32 over D=1024.
// blockIdx.z selects Q / K / V.
// Q,K out: [b][h][s][dk] bf16 (RoPE applied).  V out: [b][h][dk][s] bf16.
// ---------------------------------------------------------------------------
__global__ __launch_bounds__(256) void qkv_proj_rope(
    const __bf16* __restrict__ X,
    const __bf16* __restrict__ Wq, const __bf16* __restrict__ Wk,
    const __bf16* __restrict__ Wv,
    __bf16* __restrict__ Qo, __bf16* __restrict__ Ko, __bf16* __restrict__ Vt) {
  __shared__ __align__(16) unsigned short ldsA[2][128][32];  // 16 KB
  __shared__ __align__(16) unsigned short ldsB[2][64][32];   //  8 KB
  const int lane = threadIdx.x & 31;
  const int w    = threadIdx.x >> 5;
  const int mode = blockIdx.z;  // 0=Q, 1=K, 2=V
  const __bf16* Wmat = (mode == 0) ? Wq : (mode == 1) ? Wk : Wv;
  const int h    = blockIdx.y;                 // head index
  const int mblk = blockIdx.x * 128;
  const int m0   = mblk + (w >> 1) * 32;
  const int nw   = (w & 1) * 32;               // dk offset within head
  const __bf16* Xblk = X + (size_t)mblk * D_;
  const __bf16* Wblk = Wmat + (size_t)h * DK_ * D_;

  // Prologue: fill buffer 0.
  if (w == 0) {
    tdm_load_2d_bf16(Xblk, (unsigned)(uintptr_t)&ldsA[0][0][0],
                     D_, B_ * S_, 32, 128, D_);
    tdm_load_2d_bf16(Wblk, (unsigned)(uintptr_t)&ldsB[0][0][0],
                     D_, D_, 32, 64, D_);
    __builtin_amdgcn_s_wait_tensorcnt(0);
  }
  __syncthreads();

  v8f acc[2][2] = {};
  int buf = 0;
  for (int k0 = 0; k0 < D_; k0 += 32) {
    // Kick off next K-step tiles into the other buffer.
    if (w == 0 && (k0 + 32) < D_) {
      tdm_load_2d_bf16(Xblk + (k0 + 32), (unsigned)(uintptr_t)&ldsA[buf ^ 1][0][0],
                       D_, B_ * S_, 32, 128, D_);
      tdm_load_2d_bf16(Wblk + (k0 + 32), (unsigned)(uintptr_t)&ldsB[buf ^ 1][0][0],
                       D_, D_, 32, 64, D_);
    }

    const __bf16* Abase = reinterpret_cast<const __bf16*>(&ldsA[buf][(w >> 1) * 32][0]);
    const __bf16* Bbase = reinterpret_cast<const __bf16*>(&ldsB[buf][nw][0]);
    const v16bf a0 = load_a16x32(Abase, 32, lane);
    const v16bf a1 = load_a16x32(Abase + 16 * 32, 32, lane);
    const v16bf b0 = load_b32x16(Bbase, 32, lane);
    const v16bf b1 = load_b32x16(Bbase + 16 * 32, 32, lane);
    acc[0][0] = wmma_bf16(a0, b0, acc[0][0]);
    acc[0][1] = wmma_bf16(a0, b1, acc[0][1]);
    acc[1][0] = wmma_bf16(a1, b0, acc[1][0]);
    acc[1][1] = wmma_bf16(a1, b1, acc[1][1]);

    if (w == 0) __builtin_amdgcn_s_wait_tensorcnt(0);
    __syncthreads();
    buf ^= 1;
  }

  const int rowoff = (lane >> 4) << 3;  // C layout: VGPR v -> row v (+8 hi lanes)
  if (mode < 2) {
    __bf16* Out = (mode == 0) ? Qo : Ko;
#pragma unroll
    for (int nt = 0; nt < 2; ++nt) {
      const int dk   = nw + nt * 16 + (lane & 15);
      const float iv = __powf(10000.0f, -(float)(dk & ~1) / (float)DK_);
      const float sg = (dk & 1) ? 1.0f : -1.0f;  // even: -x2*sin; odd: +x1*sin
#pragma unroll
      for (int mt = 0; mt < 2; ++mt) {
#pragma unroll
        for (int v = 0; v < 8; ++v) {
          const int m = m0 + mt * 16 + rowoff + v;
          const int s = m & (S_ - 1);
          const int b = m >> 11;
          float sn, cs;
          __sincosf((float)s * iv, &sn, &cs);
          const float x  = acc[mt][nt][v];
          const float px = SWZF(x, 1);  // partner lane (dk ^ 1)
          const float r  = x * cs + sg * px * sn;
          Out[((size_t)(b * H_ + h) * S_ + s) * DK_ + dk] = (__bf16)r;
        }
      }
    }
  } else {
    // V: store transposed [b][h][dk][s]; rows of accum are consecutive s ->
    // pack pairs into dword stores along s.
#pragma unroll
    for (int nt = 0; nt < 2; ++nt) {
      const int dk = nw + nt * 16 + (lane & 15);
#pragma unroll
      for (int mt = 0; mt < 2; ++mt) {
        const int m = m0 + mt * 16 + rowoff;
        const int s = m & (S_ - 1);
        const int b = m >> 11;
        unsigned int* dst = reinterpret_cast<unsigned int*>(
            Vt + ((size_t)(b * H_ + h) * DK_ + dk) * S_ + s);
#pragma unroll
        for (int v = 0; v < 8; v += 2) {
          const unsigned int pk = (unsigned int)f2bf(acc[mt][nt][v]) |
                                  ((unsigned int)f2bf(acc[mt][nt][v + 1]) << 16);
          dst[v >> 1] = pk;
        }
      }
    }
  }
}

// ---------------------------------------------------------------------------
// Flash attention: 1 block = 8 independent waves, each owning 16 query rows.
// Per 32-key block: 4 WMMAs (scores) + online softmax (ds_swizzle row
// reductions) + LDS layout bounce (C-layout -> A-layout) + 4 WMMAs (PV).
// Reference semantics: clamp(scores,±80) -> causal mask -> softmax(scores/8).
// Output: att [b][s][h*64+dk] bf16 (rows ready for the output GEMM).
// ---------------------------------------------------------------------------
__global__ __launch_bounds__(256) void flash_attn(
    const __bf16* __restrict__ Q, const __bf16* __restrict__ K,
    const __bf16* __restrict__ Vt, __bf16* __restrict__ Att) {
  __shared__ unsigned short ldsP[8][16][32];  // per-wave 16x32 bf16 P tile
  const int lane = threadIdx.x & 31;
  const int w    = threadIdx.x >> 5;
  const int bh   = blockIdx.x >> 4;   // b*H + h   (S/128 = 16 q-tiles per bh)
  const int tq   = blockIdx.x & 15;
  const int q0   = tq * 128 + w * 16;
  const __bf16* Qp = Q + ((size_t)bh * S_ + q0) * DK_;
  const __bf16* Kp = K + (size_t)bh * S_ * DK_;
  const __bf16* Vp = Vt + (size_t)bh * DK_ * S_;

  const v16bf qa0 = load_a16x32(Qp, DK_, lane);
  const v16bf qa1 = load_a16x32(Qp + 32, DK_, lane);

  v8f o[4] = {};
  float mi[8], li[8];
#pragma unroll
  for (int v = 0; v < 8; ++v) { mi[v] = -3.0e38f; li[v] = 0.0f; }

  const int rowoff = (lane >> 4) << 3;
  const int c15    = lane & 15;
  const int kend   = q0 + 16;  // causal: max key needed is q0+15

  for (int kb = 0; kb < kend; kb += 32) {
    v8f s0 = {}, s1 = {};
    s0 = wmma_bf16(qa0, load_b32x16(Kp + (size_t)kb * DK_,            DK_, lane), s0);
    s0 = wmma_bf16(qa1, load_b32x16(Kp + (size_t)kb * DK_ + 32,       DK_, lane), s0);
    s1 = wmma_bf16(qa0, load_b32x16(Kp + (size_t)(kb + 16) * DK_,     DK_, lane), s1);
    s1 = wmma_bf16(qa1, load_b32x16(Kp + (size_t)(kb + 16) * DK_ + 32, DK_, lane), s1);

#pragma unroll
    for (int v = 0; v < 8; ++v) {
      const int qrow = q0 + rowoff + v;
      float a = s0[v];
      a = fminf(fmaxf(a, -80.0f), 80.0f) * 0.125f;
      if (kb + c15 > qrow) a = -3.0e38f;
      float b = s1[v];
      b = fminf(fmaxf(b, -80.0f), 80.0f) * 0.125f;
      if (kb + 16 + c15 > qrow) b = -3.0e38f;

      const float rm   = rowmax16(fmaxf(a, b));
      const float mnew = fmaxf(mi[v], rm);
      const float al   = __expf(mi[v] - mnew);
      const float p0   = __expf(a - mnew);
      const float p1   = __expf(b - mnew);
      li[v] = li[v] * al + rowsum16(p0 + p1);
      mi[v] = mnew;
      o[0][v] *= al; o[1][v] *= al; o[2][v] *= al; o[3][v] *= al;

      ldsP[w][rowoff + v][c15]      = f2bf(p0);
      ldsP[w][rowoff + v][16 + c15] = f2bf(p1);
    }
    asm volatile("" ::: "memory");  // keep LDS store->load order (in-order per wave)

    Frag16 pf;  // A-layout read of the 16x32 P tile
    pf.q[0] = *reinterpret_cast<const uint4*>(&ldsP[w][c15][rowoff]);
    pf.q[1] = *reinterpret_cast<const uint4*>(&ldsP[w][c15][rowoff + 16]);

#pragma unroll
    for (int j = 0; j < 4; ++j)
      o[j] = wmma_bf16(pf.v,
                       load_b32x16(Vp + (size_t)(j * 16) * S_ + kb, S_, lane),
                       o[j]);
  }

  const int b = bh >> 4;
  const int h = bh & 15;
#pragma unroll
  for (int j = 0; j < 4; ++j) {
    const int e = h * DK_ + j * 16 + c15;
#pragma unroll
    for (int v = 0; v < 8; ++v) {
      const int s = q0 + rowoff + v;
      Att[((size_t)b * S_ + s) * D_ + e] = (__bf16)(o[j][v] / li[v]);
    }
  }
}

// ---------------------------------------------------------------------------
// Output projection: out = att * Wo^T, fp32 result to d_out.
// (Direct-global WMMA GEMM path; compiler clauses + pipelines the loads.)
// ---------------------------------------------------------------------------
__global__ __launch_bounds__(256) void out_proj(const __bf16* __restrict__ A,
                                                const __bf16* __restrict__ Wo,
                                                float* __restrict__ Out) {
  const int lane = threadIdx.x & 31;
  const int w    = threadIdx.x >> 5;
  const int m0 = blockIdx.x * 128 + (w >> 1) * 32;
  const int n0 = blockIdx.y * 64 + (w & 1) * 32;
  const __bf16* Ap = A + (size_t)m0 * D_;
  const __bf16* Wp = Wo + (size_t)n0 * D_;

  v8f acc[2][2] = {};
  for (int k0 = 0; k0 < D_; k0 += 32) {
    __builtin_prefetch(Ap + k0 + 128, 0, 1);
    const v16bf a0 = load_a16x32(Ap + k0, D_, lane);
    const v16bf a1 = load_a16x32(Ap + 16 * D_ + k0, D_, lane);
    const v16bf b0 = load_b32x16(Wp + k0, D_, lane);
    const v16bf b1 = load_b32x16(Wp + 16 * D_ + k0, D_, lane);
    acc[0][0] = wmma_bf16(a0, b0, acc[0][0]);
    acc[0][1] = wmma_bf16(a0, b1, acc[0][1]);
    acc[1][0] = wmma_bf16(a1, b0, acc[1][0]);
    acc[1][1] = wmma_bf16(a1, b1, acc[1][1]);
  }

  const int rowoff = (lane >> 4) << 3;
#pragma unroll
  for (int mt = 0; mt < 2; ++mt) {
#pragma unroll
    for (int nt = 0; nt < 2; ++nt) {
      const int n = n0 + nt * 16 + (lane & 15);
#pragma unroll
      for (int v = 0; v < 8; ++v) {
        const int m = m0 + mt * 16 + rowoff + v;
        Out[(size_t)m * D_ + n] = acc[mt][nt][v];
      }
    }
  }
}

// ---------------------------------------------------------------------------
// Host launcher. Workspace map (bytes):
//   [0,8M)   x  bf16          [8M,10M) Wq bf16   [10M,12M) Wk bf16
//   [12M,14M) Wv bf16         [14M,16M) Wo bf16
//   [16M,24M) Q  [b,h,s,dk]   [24M,32M) K [b,h,s,dk]
//   [32M,40M) V^T [b,h,dk,s]  [40M,48M) att [b,s,d]
// ---------------------------------------------------------------------------
extern "C" void kernel_launch(void* const* d_in, const int* in_sizes, int n_in,
                              void* d_out, int out_size, void* d_ws, size_t ws_size,
                              hipStream_t stream) {
  (void)in_sizes; (void)n_in; (void)out_size; (void)ws_size;
  const float* x  = (const float*)d_in[0];
  const float* Wq = (const float*)d_in[1];
  const float* Wk = (const float*)d_in[2];
  const float* Wv = (const float*)d_in[3];
  const float* Wo = (const float*)d_in[4];

  char* ws = (char*)d_ws;
  const size_t MB = 1024u * 1024u;
  __bf16* xb  = (__bf16*)(ws + 0);
  __bf16* wqb = (__bf16*)(ws + 8 * MB);
  __bf16* wkb = (__bf16*)(ws + 10 * MB);
  __bf16* wvb = (__bf16*)(ws + 12 * MB);
  __bf16* wob = (__bf16*)(ws + 14 * MB);
  __bf16* Qb  = (__bf16*)(ws + 16 * MB);
  __bf16* Kb  = (__bf16*)(ws + 24 * MB);
  __bf16* Vtb = (__bf16*)(ws + 32 * MB);
  __bf16* Ab  = (__bf16*)(ws + 40 * MB);

  const int NX = B_ * S_ * D_;  // 4 Mi elems
  const int NW = D_ * D_;       // 1 Mi elems
  cvt_f32_bf16<<<NX / 4 / 256, 256, 0, stream>>>(x,  xb,  NX / 4);
  cvt_f32_bf16<<<NW / 4 / 256, 256, 0, stream>>>(Wq, wqb, NW / 4);
  cvt_f32_bf16<<<NW / 4 / 256, 256, 0, stream>>>(Wk, wkb, NW / 4);
  cvt_f32_bf16<<<NW / 4 / 256, 256, 0, stream>>>(Wv, wvb, NW / 4);
  cvt_f32_bf16<<<NW / 4 / 256, 256, 0, stream>>>(Wo, wob, NW / 4);

  qkv_proj_rope<<<dim3((B_ * S_) / 128, D_ / DK_, 3), 256, 0, stream>>>(
      xb, wqb, wkb, wvb, Qb, Kb, Vtb);

  flash_attn<<<B_ * H_ * (S_ / 128), 256, 0, stream>>>(Qb, Kb, Vtb, Ab);

  out_proj<<<dim3((B_ * S_) / 128, D_ / 64), 256, 0, stream>>>(Ab, wob,
                                                               (float*)d_out);
}
